// Attention_34668976013796
// MI455X (gfx1250) — matrix operand order repaired and checked by reading, not compiled, not run on hardware
//
#include <hip/hip_runtime.h>
#include <hip/hip_bf16.h>
#include <stdint.h>

// ---------------------------------------------------------------------------
// CDNA5 (gfx1250) attention block: bf16 WMMA GEMMs + fused RMSNorm epilogue +
// flash-style causal attention with transposed-score trick (wave32).
// Adds global_prefetch_b8 (via __builtin_prefetch) ahead of the WMMA loops.
// ---------------------------------------------------------------------------

typedef __bf16 bf16_t;
typedef __attribute__((ext_vector_type(16))) __bf16 v16bf;
typedef __attribute__((ext_vector_type(8)))  __bf16 v8bf;
typedef __attribute__((ext_vector_type(8)))  float  v8f;

#define N_HEADS   16
#define HEAD_DIM  128
#define D_MODEL   2048
#define EPS_RMS   1e-5f

static __device__ __forceinline__ v8f zero8() {
    v8f z = {0.f, 0.f, 0.f, 0.f, 0.f, 0.f, 0.f, 0.f};
    return z;
}

// A-operand fragment (16x32 bf16, MxK): lane row = lane&15, kbase = (lane<16)?0:16,
// halves [0..15] = K kbase..kbase+15.
// p must point at (row_base + kbase) in a K-contiguous row.
static __device__ __forceinline__ v16bf load_a_frag(const bf16_t* p) {
    v8bf lo = *(const v8bf*)p;
    v8bf hi = *(const v8bf*)(p + 8);
    return __builtin_shufflevector(lo, hi, 0, 1, 2, 3, 4, 5, 6, 7,
                                           8, 9, 10, 11, 12, 13, 14, 15);
}

static __device__ __forceinline__ v8f wmma_bf16(v16bf a, v16bf b, v8f c) {
    // D = A*B + C, fp32 accumulate
    return __builtin_amdgcn_wmma_f32_16x16x32_bf16(false, a, false, b,
                                                   (short)0, c, false, false);
}

// ---------------------------------------------------------------------------
// fp32 -> bf16 conversion
// ---------------------------------------------------------------------------
__global__ void cvt_f32_bf16(const float* __restrict__ in,
                             bf16_t* __restrict__ out, int n) {
    int i = blockIdx.x * blockDim.x + threadIdx.x;
    if (i < n) out[i] = (bf16_t)in[i];
}

// ---------------------------------------------------------------------------
// C[M,N] = A[M,K] @ Wt[N,K]^T with bf16 WMMA, fp32 accum.
// Block tile 128x128 (8 waves; wave tile 32x64 = 2x4 16x16 accumulators).
// One N tile == one head (128), so RMSNorm over head_dim fuses in-block.
//   mode 0: store fp32 row-major (final projection)
//   mode 1: per-head RMSNorm(g) then bf16 store transposed [B,H,T,hd]
//   mode 2: bf16 store transposed [B,H,hd,T]  (V for the PV matmul A-operand)
// ---------------------------------------------------------------------------
__global__ __launch_bounds__(256) void gemm_bf16_wmma(
    const bf16_t* __restrict__ A,   // [M, K]
    const bf16_t* __restrict__ Wt,  // [N, K]
    int K, int T, int mode,
    const float* __restrict__ g,
    float* __restrict__ Cf,
    bf16_t* __restrict__ Ct)
{
    __shared__ float rowsq[128];

    const int lane = threadIdx.x & 31;
    const int wave = threadIdx.x >> 5;
    const int mw   = wave & 3;            // 4 M-groups of 32 rows
    const int nw   = wave >> 2;           // 2 N-groups of 64 cols
    const int lr   = lane & 15;
    const int hi   = (lane >= 16) ? 1 : 0;
    const int kb8  = hi ? 16 : 0;

    const int rowBlock = blockIdx.x * 128;
    const int colBlock = blockIdx.y * 128;

    v8f acc[2][4];
#pragma unroll
    for (int im = 0; im < 2; ++im)
#pragma unroll
        for (int in = 0; in < 4; ++in) acc[im][in] = zero8();

    for (int k0 = 0; k0 < K; k0 += 32) {
        // Prefetch the k0+64 slabs into WGP$/L2 (global_prefetch_b8; does not
        // touch LOADcnt, so it is pure latency hiding ahead of the clauses).
        if (k0 + 64 < K) {
            __builtin_prefetch(
                A + (size_t)(rowBlock + mw * 32 + lr) * K + k0 + 64, 0, 3);
        }
        v16bf af[2], bfg[4];
#pragma unroll
        for (int im = 0; im < 2; ++im) {
            const bf16_t* pa =
                A + (size_t)(rowBlock + mw * 32 + im * 16 + lr) * K + k0 + kb8;
            af[im] = load_a_frag(pa);
        }
#pragma unroll
        for (int in = 0; in < 4; ++in) {
            // B operand: lane col = lane&15, half h <-> K = 16*hi + h
            const bf16_t* pb =
                Wt + (size_t)(colBlock + nw * 64 + in * 16 + lr) * K + k0 + hi * 16;
            bfg[in] = *(const v16bf*)pb;
        }
        if (k0 + 64 < K) {
            __builtin_prefetch(
                Wt + (size_t)(colBlock + nw * 64 + lr) * K + k0 + 64, 0, 3);
        }
#pragma unroll
        for (int im = 0; im < 2; ++im)
#pragma unroll
            for (int in = 0; in < 4; ++in)
                acc[im][in] = wmma_bf16(af[im], bfg[in], acc[im][in]);
    }

    if (mode == 1) {
        // per-row (token) sum of squares over the full head (128 cols)
        if (threadIdx.x < 128) rowsq[threadIdx.x] = 0.f;
        __syncthreads();
#pragma unroll
        for (int im = 0; im < 2; ++im) {
#pragma unroll
            for (int v = 0; v < 8; ++v) {
                float s = 0.f;
#pragma unroll
                for (int in = 0; in < 4; ++in) {
                    float x = acc[im][in][v];
                    s += x * x;
                }
                s += __shfl_xor(s, 1, 32);
                s += __shfl_xor(s, 2, 32);
                s += __shfl_xor(s, 4, 32);
                s += __shfl_xor(s, 8, 32);
                if (lr == 0)
                    atomicAdd(&rowsq[mw * 32 + im * 16 + v + 8 * hi], s);
            }
        }
        __syncthreads();
    }

    const int h = blockIdx.y;  // head index for modes 1/2 (N tile == head)
#pragma unroll
    for (int im = 0; im < 2; ++im)
#pragma unroll
        for (int in = 0; in < 4; ++in)
#pragma unroll
            for (int v = 0; v < 8; ++v) {
                const int r  = mw * 32 + im * 16 + v + 8 * hi;  // 0..127
                const int c  = nw * 64 + in * 16 + lr;          // 0..127
                const int tg = rowBlock + r;
                float val = acc[im][in][v];
                if (mode == 0) {
                    Cf[(size_t)tg * (size_t)(gridDim.y * 128) + colBlock + c] = val;
                } else {
                    const int b = tg / T, t = tg - b * T;
                    if (mode == 1) {
                        float ms = rowsq[r] * (1.0f / HEAD_DIM);
                        val = val * rsqrtf(ms + EPS_RMS) * g[c];
                        Ct[(((size_t)(b * N_HEADS + h)) * T + t) * HEAD_DIM + c] =
                            (bf16_t)val;
                    } else {
                        Ct[(((size_t)(b * N_HEADS + h)) * HEAD_DIM + c) * T + t] =
                            (bf16_t)val;
                    }
                }
            }
}

// ---------------------------------------------------------------------------
// Flash-style causal attention, one 16-row Q tile per wave.
// Scores computed transposed: St = K_blk(16x d) x Q^T(d x 16q), so each lane
// owns one query column -> softmax stats need a single shfl_xor(16).
// PV matmul computed transposed: out^T(d x q) = V^T(d x k) x P^T(k x q),
// where P^T already sits lane-aligned with the B-operand layout (16 shuffles).
// ---------------------------------------------------------------------------
__global__ __launch_bounds__(256) void attn_fwd_wmma(
    const bf16_t* __restrict__ Qn,  // [B,H,T,hd] (rms-normed)
    const bf16_t* __restrict__ Kn,  // [B,H,T,hd] (rms-normed)
    const bf16_t* __restrict__ Vt,  // [B,H,hd,T]
    bf16_t* __restrict__ O,         // [B,T,H*hd]
    int T)
{
    const int HD   = HEAD_DIM;
    const int lane = threadIdx.x & 31;
    const int wave = threadIdx.x >> 5;
    const int bh   = blockIdx.x;           // b*H + h
    const int b    = bh >> 4, h = bh & 15;
    const int qbase = (blockIdx.y * 8 + wave) * 16;
    const int lq   = lane & 15;
    const int hi   = (lane >= 16) ? 1 : 0;
    const int kb8  = hi ? 16 : 0;
    const int qg   = qbase + lq;

    const bf16_t* Qh = Qn + (size_t)bh * T * HD;
    const bf16_t* Kh = Kn + (size_t)bh * T * HD;
    const bf16_t* Vh = Vt + (size_t)bh * HD * T;

    // Q as B operand (d x q), 4 d-chunks of 32: lane col = q, half h <-> d
    v16bf qb[4];
#pragma unroll
    for (int c = 0; c < 4; ++c)
        qb[c] = *(const v16bf*)(Qh + (size_t)qg * HD + c * 32 + hi * 16);

    v8f acc[8];  // out^T accumulators: 8 d-tiles x (16d x 16q)
#pragma unroll
    for (int dt = 0; dt < 8; ++dt) acc[dt] = zero8();

    float m = -3.0e38f, l = 0.f;
    const float sc = 0.08838834764831845f;  // 1/sqrt(128)

    const int nkb = (qbase + 16 + 31) >> 5;  // key blocks of 32 up to diagonal
    for (int kb = 0; kb < nkb; ++kb) {
        const int k0 = kb * 32;

        // Prefetch next key block's K rows and V rows (global_prefetch_b8).
        if (kb + 1 < nkb) {
            __builtin_prefetch(Kh + (size_t)(k0 + 32 + lq) * HD, 0, 3);
            __builtin_prefetch(Kh + (size_t)(k0 + 48 + lq) * HD, 0, 3);
            __builtin_prefetch(Vh + (size_t)(lq * 8 + hi * 4) * T + k0 + 32, 0, 3);
        }

        v8f st0 = zero8(), st1 = zero8();  // St: keys(M) x queries(N)
#pragma unroll
        for (int c = 0; c < 4; ++c) {
            const bf16_t* p0 = Kh + (size_t)(k0 + lq) * HD + c * 32 + kb8;
            v16bf ka0 = load_a_frag(p0);            // keys k0..k0+15
            v16bf ka1 = load_a_frag(p0 + 16 * HD);  // keys k0+16..k0+31
            st0 = wmma_bf16(ka0, qb[c], st0);
            st1 = wmma_bf16(ka1, qb[c], st1);
        }

        const bool needMask = (k0 + 31) > qbase;
#pragma unroll
        for (int v = 0; v < 8; ++v) {
            float a0 = st0[v] * sc, a1 = st1[v] * sc;
            if (needMask) {
                int kt0 = k0 + v + 8 * hi;       // key row of st0[v]
                if (kt0 > qg)      a0 = -3.0e38f;
                if (kt0 + 16 > qg) a1 = -3.0e38f;
            }
            st0[v] = a0;
            st1[v] = a1;
        }

        // online softmax per query column (lane pair L, L+16 share a column)
        float bm = -3.0e38f;
#pragma unroll
        for (int v = 0; v < 8; ++v) bm = fmaxf(bm, fmaxf(st0[v], st1[v]));
        bm = fmaxf(bm, __shfl_xor(bm, 16, 32));
        float mn    = fmaxf(m, bm);
        float alpha = __expf(m - mn);
        float ls    = 0.f;
#pragma unroll
        for (int v = 0; v < 8; ++v) {
            float p0 = __expf(st0[v] - mn);
            float p1 = __expf(st1[v] - mn);
            st0[v] = p0;
            st1[v] = p1;
            ls += p0 + p1;
        }
        ls += __shfl_xor(ls, 16, 32);
        l = l * alpha + ls;
        m = mn;
#pragma unroll
        for (int dt = 0; dt < 8; ++dt) acc[dt] *= alpha;

        // Repack P^T (C layout) into B-operand layout: half v <-> key 16*hi+v.
        // lanes<16 need keys 8..15 from partner st0; lanes>=16 need keys
        // 16..23 from partner st1.
        v16bf pb;
#pragma unroll
        for (int v = 0; v < 8; ++v) {
            float o0 = __shfl_xor(st0[v], 16, 32);
            float o1 = __shfl_xor(st1[v], 16, 32);
            float lo = hi ? o1 : st0[v];
            float hv = hi ? st1[v] : o0;
            pb[v]     = (bf16_t)lo;
            pb[v + 8] = (bf16_t)hv;
        }

        // out^T += V^T(d x k) * P^T(k x q)
#pragma unroll
        for (int dt = 0; dt < 8; ++dt) {
            const bf16_t* pv = Vh + (size_t)(dt * 16 + lq) * T + k0 + kb8;
            v16bf va = load_a_frag(pv);
            acc[dt] = wmma_bf16(va, pb, acc[dt]);
        }
    }

    const float inv = 1.0f / l;
    bf16_t* orow = O + ((size_t)b * T + qg) * (size_t)D_MODEL + h * HD;
#pragma unroll
    for (int dt = 0; dt < 8; ++dt)
#pragma unroll
        for (int v = 0; v < 8; ++v)
            orow[dt * 16 + v + 8 * hi] = (bf16_t)(acc[dt][v] * inv);
}

// ---------------------------------------------------------------------------
extern "C" void kernel_launch(void* const* d_in, const int* in_sizes, int n_in,
                              void* d_out, int out_size, void* d_ws, size_t ws_size,
                              hipStream_t stream) {
    const float* x  = (const float*)d_in[0];
    const float* Wq = (const float*)d_in[1];
    const float* Wk = (const float*)d_in[2];
    const float* Wv = (const float*)d_in[3];
    const float* Wo = (const float*)d_in[4];
    const float* gq = (const float*)d_in[5];
    const float* gk = (const float*)d_in[6];

    const int T = 2048;
    const int M = in_sizes[0] / D_MODEL;           // B*T = 8192
    const size_t xN = (size_t)M * D_MODEL;         // 16,777,216
    const size_t wN = (size_t)D_MODEL * D_MODEL;   // 4,194,304

    char* ws = (char*)d_ws;
    bf16_t* xb  = (bf16_t*)ws; ws += xN * 2;
    bf16_t* wqb = (bf16_t*)ws; ws += wN * 2;
    bf16_t* wkb = (bf16_t*)ws; ws += wN * 2;
    bf16_t* wvb = (bf16_t*)ws; ws += wN * 2;
    bf16_t* wob = (bf16_t*)ws; ws += wN * 2;
    bf16_t* Qn  = (bf16_t*)ws; ws += xN * 2;
    bf16_t* Kn  = (bf16_t*)ws; ws += xN * 2;
    bf16_t* Vt  = (bf16_t*)ws; ws += xN * 2;
    bf16_t* Ao  = (bf16_t*)ws; ws += xN * 2;

    cvt_f32_bf16<<<(int)((xN + 255) / 256), 256, 0, stream>>>(x, xb, (int)xN);
    cvt_f32_bf16<<<(int)((wN + 255) / 256), 256, 0, stream>>>(Wq, wqb, (int)wN);
    cvt_f32_bf16<<<(int)((wN + 255) / 256), 256, 0, stream>>>(Wk, wkb, (int)wN);
    cvt_f32_bf16<<<(int)((wN + 255) / 256), 256, 0, stream>>>(Wv, wvb, (int)wN);
    cvt_f32_bf16<<<(int)((wN + 255) / 256), 256, 0, stream>>>(Wo, wob, (int)wN);

    dim3 gg(M / 128, D_MODEL / 128);  // 64 x 16 blocks, 256 threads
    gemm_bf16_wmma<<<gg, 256, 0, stream>>>(xb, wqb, D_MODEL, T, 1, gq, nullptr, Qn);
    gemm_bf16_wmma<<<gg, 256, 0, stream>>>(xb, wkb, D_MODEL, T, 1, gk, nullptr, Kn);
    gemm_bf16_wmma<<<gg, 256, 0, stream>>>(xb, wvb, D_MODEL, T, 2, nullptr, nullptr, Vt);

    dim3 ga((M / T) * N_HEADS, T / 128);  // 64 x 16 blocks, 8 q-tiles per block
    attn_fwd_wmma<<<ga, 256, 0, stream>>>(Qn, Kn, Vt, Ao, T);

    gemm_bf16_wmma<<<gg, 256, 0, stream>>>(Ao, wob, D_MODEL, T, 0, nullptr,
                                           (float*)d_out, nullptr);
}
